// GraphRewiring_61624190763587
// MI455X (gfx1250) — 8-bit matrix operand order repaired and checked by reading, compiled, not run on hardware
//
#include <hip/hip_runtime.h>
#include <hip/hip_bf16.h>
#include <stdint.h>

typedef int v8i __attribute__((ext_vector_type(8)));

#define NND 1024
#define NNSQ (NND * NND)

__global__ void zero_u32_kernel(uint32_t* __restrict__ p, int n_dwords) {
    int i = blockIdx.x * blockDim.x + threadIdx.x;
    if (i < n_dwords) p[i] = 0u;
}

__global__ void scatter_kernel(const int* __restrict__ ei,
                               const float* __restrict__ attr,
                               uint8_t* __restrict__ Arow, int E) {
    int e = blockIdx.x * blockDim.x + threadIdx.x;
    if (e >= E) return;
    if (attr[e] == 0.0f) {
        int s = ei[e];
        int d = ei[E + e];
        Arow[(size_t)s * NND + d] = (uint8_t)1;
    }
}

__global__ void seed_kernel(const uint8_t* __restrict__ Arow,
                            uint8_t* __restrict__ Mrow,
                            uint8_t* __restrict__ Mcol) {
    int p = blockIdx.x * blockDim.x + threadIdx.x;
    if (p >= NNSQ) return;
    int i = p >> 10, j = p & (NND - 1);
    uint8_t v = (uint8_t)(Arow[p] | (uint8_t)(i == j));
    Mrow[p] = v;
    Mcol[(size_t)j * NND + i] = v;
}

__global__ __launch_bounds__(256)
void bmm_iu8_kernel(const uint8_t* __restrict__ Srow,
                    const uint8_t* __restrict__ Scol,
                    uint8_t* __restrict__ Crow,
                    uint8_t* __restrict__ Ccol) {
    const int lane = threadIdx.x & 31;
    const int wave = threadIdx.x >> 5;
    const int tile = blockIdx.x * 8 + wave;
    const int i0 = (tile >> 6) << 4;
    const int j0 = (tile & 63) << 4;
    const int m  = lane & 15;
    const int h  = lane >> 4;

    const uint8_t* arow = Srow + (size_t)(i0 + m) * NND;
    const uint8_t* bcol = Scol + (size_t)(j0 + m) * NND;

    v8i c = {};
    for (int kt = 0; kt < 16; ++kt) {
        const uint32_t* a32 = (const uint32_t*)(arow + kt * 64);
        const uint32_t* b32 = (const uint32_t*)(bcol + kt * 64);
        v8i a, b;
        const int d0 = h * 2;
        a[0] = a32[d0 + 0];  a[1] = a32[d0 + 1];
        a[2] = a32[d0 + 4];  a[3] = a32[d0 + 5];
        a[4] = a32[d0 + 8];  a[5] = a32[d0 + 9];
        a[6] = a32[d0 + 12]; a[7] = a32[d0 + 13];
        const int e0 = h * 2;
        b[0] = b32[e0 + 0];  b[1] = b32[e0 + 1];
        b[2] = b32[e0 + 4];  b[3] = b32[e0 + 5];
        b[4] = b32[e0 + 8];  b[5] = b32[e0 + 9];
        b[6] = b32[e0 + 12]; b[7] = b32[e0 + 13];
        c = __builtin_amdgcn_wmma_i32_16x16x64_iu8(false, a, false, b, c,
                                                   false, false);
    }

    uint32_t lo = 0, hi = 0;
#pragma unroll
    for (int r = 0; r < 4; ++r) lo |= (c[r]     != 0 ? 1u : 0u) << (8 * r);
#pragma unroll
    for (int r = 0; r < 4; ++r) hi |= (c[4 + r] != 0 ? 1u : 0u) << (8 * r);

    const int col = j0 + m;
    uint32_t* cc = (uint32_t*)(Ccol + (size_t)col * NND + i0 + h * 8);
    cc[0] = lo;
    cc[1] = hi;
#pragma unroll
    for (int r = 0; r < 8; ++r)
        Crow[(size_t)(i0 + h * 8 + r) * NND + col] = (uint8_t)(c[r] != 0);
}

__global__ void emit_edges_kernel(const int* __restrict__ ei,
                                  const float* __restrict__ attr,
                                  float* __restrict__ out, int E, int EN) {
    int e = blockIdx.x * blockDim.x + threadIdx.x;
    if (e >= E) return;
    out[e]          = (float)ei[e];
    out[EN + e]     = (float)ei[E + e];
    out[2 * EN + e] = attr[e];
}

__global__ void emit_new_kernel(const uint8_t* __restrict__ T,
                                const uint8_t* __restrict__ Arow,
                                float* __restrict__ out, int E, int EN) {
    int p = blockIdx.x * blockDim.x + threadIdx.x;
    if (p >= NNSQ) return;
    bool f = (T[p] != 0) && (Arow[p] == 0);
    int i = p >> 10, j = p & (NND - 1);
    out[E + p]          = f ? (float)i : 0.0f;
    out[EN + E + p]     = f ? (float)j : 0.0f;
    out[2 * EN + E + p] = f ? 3.0f : -1.0f;
}

extern "C" void kernel_launch(void* const* d_in, const int* in_sizes, int n_in,
                              void* d_out, int out_size, void* d_ws, size_t ws_size,
                              hipStream_t stream) {
    const int*   ei   = (const int*)d_in[0];
    const float* attr = (const float*)d_in[1];
    const int E  = in_sizes[0] / 2;
    const int EN = E + NNSQ;
    float* out = (float*)d_out;

    uint8_t* ws = (uint8_t*)d_ws;
    const size_t MB = (size_t)1 << 20;
    uint8_t* Arow    = ws;
    uint8_t* Mrow[2] = { ws + 1 * MB, ws + 3 * MB };
    uint8_t* Mcol[2] = { ws + 2 * MB, ws + 4 * MB };
    uint8_t* Trow    = ws + 5 * MB;

    zero_u32_kernel<<<(NNSQ / 4 + 255) / 256, 256, 0, stream>>>(
        (uint32_t*)Arow, NNSQ / 4);
    scatter_kernel<<<(E + 255) / 256, 256, 0, stream>>>(ei, attr, Arow, E);

    seed_kernel<<<NNSQ / 256, 256, 0, stream>>>(Arow, Mrow[0], Mcol[0]);

    int cur = 0;
    for (int it = 0; it < 10; ++it) {
        bmm_iu8_kernel<<<512, 256, 0, stream>>>(Mrow[cur], Mcol[cur],
                                                Mrow[cur ^ 1], Mcol[cur ^ 1]);
        cur ^= 1;
    }

    bmm_iu8_kernel<<<512, 256, 0, stream>>>(Arow, Mcol[cur], Trow,
                                            Mcol[cur ^ 1]);

    emit_edges_kernel<<<(E + 255) / 256, 256, 0, stream>>>(ei, attr, out, E, EN);
    emit_new_kernel<<<NNSQ / 256, 256, 0, stream>>>(Trow, Arow, out, E, EN);
}
